// FIN_b_89859305767637
// MI455X (gfx1250) — compile-verified, matrix operand order checked by reading, not run on hardware
//
#include <hip/hip_runtime.h>
#include <hip/hip_bf16.h>

typedef __attribute__((ext_vector_type(16))) _Float16 v16h;
typedef __attribute__((ext_vector_type(8)))  _Float16 v8h;
typedef __attribute__((ext_vector_type(8)))  float    v8f;

#define B_ROWS   2048
#define NDF      512
#define NCF      256
#define NH0      1024      // NDF + 2*NCC
#define CHANNEL  1017
#define FIELD    8
#define EMB      16
#define HID      1020
#define RES_LD   2048      // padded K for GEMM3 (2041 -> 2048)
#define X_LD     1024
#define H_LD     1024
#define W1T_N    1024      // padded N for GEMM3 (1020 -> 1024)

// ---------------- prep kernels ----------------

__global__ void k_cvt_f16(const float* __restrict__ src, _Float16* __restrict__ dst, int n) {
    int i = blockIdx.x * blockDim.x + threadIdx.x;
    if (i < n) dst[i] = (_Float16)src[i];
}

// dst[n * dstK + k] = src[k * srcN + n], zero-padded to (dstN, dstK)
__global__ void k_transpose_f16(const float* __restrict__ src, int srcK, int srcN,
                                _Float16* __restrict__ dst, int dstK, int dstN) {
    int k = blockIdx.x * blockDim.x + threadIdx.x;
    int n = blockIdx.y * blockDim.y + threadIdx.y;
    if (k >= dstK || n >= dstN) return;
    float val = (k < srcK && n < srcN) ? src[(size_t)k * srcN + n] : 0.f;
    dst[(size_t)n * dstK + k] = (_Float16)val;
}

// zero res columns 2041..2047 (GEMM3 K-padding)
__global__ void k_zero_res_tail(_Float16* __restrict__ res) {
    int i = blockIdx.x * blockDim.x + threadIdx.x;   // B_ROWS * 7
    if (i >= B_ROWS * 7) return;
    int r = i / 7, c = 2041 + (i % 7);
    res[(size_t)r * RES_LD + c] = (_Float16)0.f;
}

// ---------------- WMMA GEMM ----------------
// C[M,N] = act(A[M,K] @ BT[N,K]^T + bias)
// One wave computes a 32x32 output tile (2x2 grid of 16x16 WMMA tiles):
// 4 WMMAs per K-step for 4 fragment loads -> 2x arithmetic intensity vs 1-tile waves.
// 8 waves/block arranged 2(M) x 4(N) -> block tile 64 x 128.

__device__ __forceinline__ v16h load_frag(const _Float16* p) {
    v8h lo = *reinterpret_cast<const v8h*>(p);
    v8h hi = *reinterpret_cast<const v8h*>(p + 16);
    v16h r;
#pragma unroll
    for (int i = 0; i < 8; ++i) { r[i] = lo[i]; r[i + 8] = hi[i]; }
    return r;
}

template <int ACT>  // 0=none 1=relu 2=leaky(0.01)
__global__ __launch_bounds__(256)
void k_gemm_wmma(const _Float16* __restrict__ A, int lda,
                 const _Float16* __restrict__ BT, int ldb,
                 const float* __restrict__ bias, int nBias,
                 float* __restrict__ outF32, int ldo32,
                 _Float16* __restrict__ outF16, int ldo16, int colOff,
                 int K) {
    const int lane  = threadIdx.x & 31;
    const int wave  = threadIdx.x >> 5;
    const int tileM = blockIdx.y * 64  + (wave & 1) * 32;
    const int tileN = blockIdx.x * 128 + (wave >> 1) * 32;
    const int grp   = lane >> 4;   // 0/1
    const int mr    = lane & 15;

    // A frag: lane(row m=mr) holds K = grp*8 + [0..7] and +16 (16-bit A layout, ISA 7.12.2)
    const _Float16* pA0 = A  + (size_t)(tileM + mr) * lda + grp * 8;
    const _Float16* pA1 = pA0 + (size_t)16 * lda;
    // B frag: mirrored layout, lane holds row n=mr of BT (= column of B)
    const _Float16* pB0 = BT + (size_t)(tileN + mr) * ldb + grp * 8;
    const _Float16* pB1 = pB0 + (size_t)16 * ldb;

    v8f acc00 = {}, acc01 = {}, acc10 = {}, acc11 = {};
    for (int k0 = 0; k0 < K; k0 += 32) {
        v16h a0 = load_frag(pA0 + k0);
        v16h a1 = load_frag(pA1 + k0);
        v16h b0 = load_frag(pB0 + k0);
        v16h b1 = load_frag(pB1 + k0);
        acc00 = __builtin_amdgcn_wmma_f32_16x16x32_f16(false, a0, false, b0, (short)0, acc00, false, false);
        acc01 = __builtin_amdgcn_wmma_f32_16x16x32_f16(false, a0, false, b1, (short)0, acc01, false, false);
        acc10 = __builtin_amdgcn_wmma_f32_16x16x32_f16(false, a1, false, b0, (short)0, acc10, false, false);
        acc11 = __builtin_amdgcn_wmma_f32_16x16x32_f16(false, a1, false, b1, (short)0, acc11, false, false);
    }

    // C layout: lane L -> N = L%16, M = vgpr + 8*(L/16)
    const int n0 = tileN + mr;
    const int n1 = tileN + 16 + mr;
    float bn0 = (bias != nullptr && n0 < nBias) ? bias[n0] : 0.f;
    float bn1 = (bias != nullptr && n1 < nBias) ? bias[n1] : 0.f;

#pragma unroll
    for (int mt = 0; mt < 2; ++mt) {
        const v8f& aN0 = mt ? acc10 : acc00;
        const v8f& aN1 = mt ? acc11 : acc01;
#pragma unroll
        for (int g = 0; g < 8; ++g) {
            int m = tileM + mt * 16 + g + grp * 8;
            float v0 = aN0[g] + bn0;
            float v1 = aN1[g] + bn1;
            if (ACT == 1) { v0 = v0 > 0.f ? v0 : 0.f;        v1 = v1 > 0.f ? v1 : 0.f; }
            if (ACT == 2) { v0 = v0 > 0.f ? v0 : 0.01f * v0; v1 = v1 > 0.f ? v1 : 0.01f * v1; }
            if (outF32) {
                outF32[(size_t)m * ldo32 + colOff + n0] = v0;
                outF32[(size_t)m * ldo32 + colOff + n1] = v1;
            }
            if (outF16) {
                outF16[(size_t)m * ldo16 + colOff + n0] = (_Float16)v0;
                outF16[(size_t)m * ldo16 + colOff + n1] = (_Float16)v1;
            }
        }
    }
}

// ---------------- FM second-order + linear term ----------------
// res[b, 1024+c] = lin_b + x[b,c:c+8]@lin_w + 0.5*sum_e((sum_f xv)^2 - sum_f (xv)^2)

__global__ __launch_bounds__(256)
void k_fm(const float* __restrict__ x, const float* __restrict__ v,
          const float* __restrict__ lin_w, const float* __restrict__ lin_b,
          _Float16* __restrict__ res) {
    const int c = blockIdx.y;                 // 0..CHANNEL-1
    __shared__ float sv[FIELD][EMB];
    __shared__ float slw[FIELD];
    int t = threadIdx.x;
    if (t < FIELD * EMB) sv[t / EMB][t % EMB] = v[(size_t)c * FIELD * EMB + t];
    if (t < FIELD)       slw[t] = lin_w[t];
    __syncthreads();

    int b = blockIdx.x * blockDim.x + t;      // 0..B_ROWS-1
    if (b >= B_ROWS) return;

    float xf[FIELD];
    float lin = lin_b[0];
#pragma unroll
    for (int f = 0; f < FIELD; ++f) {
        xf[f] = x[(size_t)b * X_LD + c + f];
        lin += xf[f] * slw[f];
    }
    float fm = 0.f;
#pragma unroll
    for (int e = 0; e < EMB; ++e) {
        float s = 0.f, q = 0.f;
#pragma unroll
        for (int f = 0; f < FIELD; ++f) {
            float p = xf[f] * sv[f][e];
            s += p; q += p * p;
        }
        fm += s * s - q;
    }
    res[(size_t)b * RES_LD + NH0 + c] = (_Float16)(lin + 0.5f * fm);
}

// ---------------- head: pred = sigmoid(h @ W2 + b2) ----------------

__global__ __launch_bounds__(256)
void k_head(const float* __restrict__ h, const float* __restrict__ W2,
            const float* __restrict__ b2, float* __restrict__ out) {
    int b = blockIdx.x, t = threadIdx.x;
    float p = 0.f;
    for (int j = t; j < HID; j += 256) p += h[(size_t)b * H_LD + j] * W2[j];
    __shared__ float sb[256];
    sb[t] = p;
    __syncthreads();
    for (int s = 128; s > 0; s >>= 1) {
        if (t < s) sb[t] += sb[t + s];
        __syncthreads();
    }
    if (t == 0) out[b] = 1.f / (1.f + __expf(-(sb[0] + b2[0])));
}

// ---------------- launcher ----------------

extern "C" void kernel_launch(void* const* d_in, const int* in_sizes, int n_in,
                              void* d_out, int out_size, void* d_ws, size_t ws_size,
                              hipStream_t stream) {
    (void)in_sizes; (void)n_in; (void)out_size; (void)ws_size;
    const float* dx    = (const float*)d_in[0];
    const float* cx    = (const float*)d_in[1];
    const float* W_d   = (const float*)d_in[2];
    const float* b_d   = (const float*)d_in[3];
    const float* W_c   = (const float*)d_in[4];
    const float* b_c   = (const float*)d_in[5];
    const float* v     = (const float*)d_in[6];
    const float* lin_w = (const float*)d_in[7];
    const float* lin_b = (const float*)d_in[8];
    const float* W1    = (const float*)d_in[9];
    const float* b1    = (const float*)d_in[10];
    const float* W2    = (const float*)d_in[11];
    const float* b2    = (const float*)d_in[12];
    float* pred = (float*)d_out;

    char* ws = (char*)d_ws;
    size_t off = 0;
    auto alloc = [&](size_t bytes) -> void* {
        void* p = ws + off;
        off = (off + bytes + 255) & ~(size_t)255;
        return p;
    };
    _Float16* A1   = (_Float16*)alloc((size_t)B_ROWS * NDF * 2);     // dx f16
    _Float16* A2   = (_Float16*)alloc((size_t)B_ROWS * NCF * 2);     // cx f16
    _Float16* WdT  = (_Float16*)alloc((size_t)512 * 512 * 2);        // [N=512,K=512]
    _Float16* WcT  = (_Float16*)alloc((size_t)512 * 256 * 2);        // [N=512,K=256]
    _Float16* W1T  = (_Float16*)alloc((size_t)W1T_N * RES_LD * 2);   // [N=1024,K=2048] padded
    float*    xF32 = (float*)   alloc((size_t)B_ROWS * X_LD * 4);
    _Float16* res  = (_Float16*)alloc((size_t)B_ROWS * RES_LD * 2);
    float*    hF32 = (float*)   alloc((size_t)B_ROWS * H_LD * 4);

    // --- prep ---
    {
        int n1 = B_ROWS * NDF;
        k_cvt_f16<<<(n1 + 255) / 256, 256, 0, stream>>>(dx, A1, n1);
        int n2 = B_ROWS * NCF;
        k_cvt_f16<<<(n2 + 255) / 256, 256, 0, stream>>>(cx, A2, n2);
    }
    {
        dim3 blk(16, 16);
        k_transpose_f16<<<dim3(512 / 16, 512 / 16), blk, 0, stream>>>(W_d, 512, 512, WdT, 512, 512);
        k_transpose_f16<<<dim3(256 / 16, 512 / 16), blk, 0, stream>>>(W_c, 256, 512, WcT, 256, 512);
        k_transpose_f16<<<dim3(RES_LD / 16, W1T_N / 16), blk, 0, stream>>>(W1, NH0 + CHANNEL, HID, W1T, RES_LD, W1T_N);
    }
    k_zero_res_tail<<<(B_ROWS * 7 + 255) / 256, 256, 0, stream>>>(res);

    // --- GEMM1: x_d = relu(dx @ W_d + b_d) -> x[:,0:512] (f32) and res[:,0:512] (f16)
    k_gemm_wmma<1><<<dim3(512 / 128, B_ROWS / 64), 256, 0, stream>>>(
        A1, NDF, WdT, NDF, b_d, 512, xF32, X_LD, res, RES_LD, 0, NDF);

    // --- GEMM2: x_c = relu(cx @ W_c + b_c) -> x[:,512:1024], res[:,512:1024]
    k_gemm_wmma<1><<<dim3(512 / 128, B_ROWS / 64), 256, 0, stream>>>(
        A2, NCF, WcT, NCF, b_c, 512, xF32 + 512, X_LD, res + 512, RES_LD, 0, NCF);

    // --- FM: res[:,1024:2041]
    k_fm<<<dim3(B_ROWS / 256, CHANNEL), 256, 0, stream>>>(xF32, v, lin_w, lin_b, res);

    // --- GEMM3: h = leaky_relu(res @ W1 + b1)
    k_gemm_wmma<2><<<dim3(W1T_N / 128, B_ROWS / 64), 256, 0, stream>>>(
        res, RES_LD, W1T, RES_LD, b1, HID, hF32, H_LD, (_Float16*)nullptr, 0, 0, RES_LD);

    // --- head: pred = sigmoid(h @ W2 + b2)
    k_head<<<B_ROWS, 256, 0, stream>>>(hF32, W2, b2, pred);
}
